// RPN_21036749816355
// MI455X (gfx1250) — matrix operand order repaired and checked by reading, not compiled, not run on hardware
//
#include <hip/hip_runtime.h>

typedef __attribute__((ext_vector_type(16))) _Float16 v16h;
typedef __attribute__((ext_vector_type(8)))  _Float16 v8h;
typedef __attribute__((ext_vector_type(8)))  float    v8f;

static constexpr int BB = 2;     // batch
static constexpr int NN = 8192;  // points
static constexpr int KK = 32;    // neighbors per group

static inline int cp16(int x){ return (x + 15) & ~15; }
static inline int cp32(int x){ return (x + 31) & ~31; }
static inline unsigned nblk(size_t n, int t){ return (unsigned)((n + (size_t)t - 1) / (size_t)t); }

// ---------------------------------------------------------------------------
// WMMA GEMM:  Y[rows,Opad](f32) = Xh[rows,Cpad](f16) * Wh[Opad,Cpad](f16)^T + bias
// rows % 16 == 0, Cpad % 32 == 0, Opad % 16 == 0.
// 128-thread block = 4 waves; wave w handles M-tile (blockIdx.x*4 + w).
// Inner loop: 4 x b128 loads + 1 x v_wmma per 32-deep K step.
// ---------------------------------------------------------------------------
__global__ void k_gemm(const _Float16* __restrict__ Xh, const _Float16* __restrict__ Wh,
                       const float* __restrict__ bias, float* __restrict__ Y,
                       int rows, int Cpad, int O, int Opad)
{
  int wid  = threadIdx.x >> 5;
  int lane = threadIdx.x & 31;
  int tm = blockIdx.x * 4 + wid;
  if (tm * 16 >= rows) return;                       // wave-uniform guard (EXEC all-ones)
  int to = blockIdx.y;
  int m = lane & 15, h = lane >> 4;
  int rowBase = tm * 16;
  int nG = to * 16 + m;
  // A layout (16-bit 16x32): lanes 0-15 hold K 0-7 / 16-23 (+8h for lanes 16-31)
  const _Float16* xr = Xh + (size_t)(rowBase + m) * Cpad + 8 * h;
  // B layout: lane = column n, same K order as A: K 0-7 / 16-23 (+8h)
  const _Float16* wr = Wh + (size_t)nG * Cpad + 8 * h;
  v8f acc = {0.f,0.f,0.f,0.f,0.f,0.f,0.f,0.f};
  for (int k = 0; k < Cpad; k += 32) {
    v8h x0 = *(const v8h*)(xr + k);        // K = k+8h+0..7
    v8h x1 = *(const v8h*)(xr + k + 16);   // K = k+8h+16..23
    v8h w0 = *(const v8h*)(wr + k);        // K = k+8h+0..7
    v8h w1 = *(const v8h*)(wr + k + 16);   // K = k+8h+16..23
    v16h a, b;
#pragma unroll
    for (int i = 0; i < 8; ++i) {
      a[i] = x0[i]; a[i + 8] = x1[i];
      b[i] = w0[i]; b[i + 8] = w1[i];
    }
    acc = __builtin_amdgcn_wmma_f32_16x16x32_f16(false, a, false, b, (short)0, acc,
                                                 false, false);
  }
  float bv = (nG < O) ? bias[nG] : 0.f;
#pragma unroll
  for (int r = 0; r < 8; ++r) {                      // D: M = r + 8h, N = lane&15
    int row = rowBase + r + 8 * h;
    Y[(size_t)row * Opad + nG] = (nG < O) ? (acc[r] + bv) : 0.f;
  }
}

// f32 -> f16 streaming convert (activations, once per layer)
__global__ void k_cvt_h(const float* __restrict__ X, _Float16* __restrict__ Xh, int total)
{
  int i = blockIdx.x * blockDim.x + threadIdx.x;
  if (i >= total) return;
  Xh[i] = (_Float16)X[i];
}

// f32 [O,C] -> f16 [Opad,Cpad], zero padded
__global__ void k_wcvt(const float* __restrict__ W, _Float16* __restrict__ Wh,
                       int O, int C, int Opad, int Cpad)
{
  int i = blockIdx.x * blockDim.x + threadIdx.x;
  if (i >= Opad * Cpad) return;
  int o = i / Cpad, c = i % Cpad;
  Wh[i] = (o < O && c < C) ? (_Float16)W[(size_t)o * C + c] : (_Float16)0.f;
}

// per-channel sum / sumsq over rows (one block per channel)
__global__ void k_bnstat(const float* __restrict__ Y, int rows, int Opad,
                         float* __restrict__ sums, float* __restrict__ sqs, int O)
{
  int c = blockIdx.x; if (c >= O) return;
  __shared__ float ss[256], sq[256];
  float s = 0.f, q = 0.f;
  for (int r = threadIdx.x; r < rows; r += blockDim.x) {
    float v = Y[(size_t)r * Opad + c]; s += v; q += v * v;
  }
  ss[threadIdx.x] = s; sq[threadIdx.x] = q; __syncthreads();
  for (int w = 128; w > 0; w >>= 1) {
    if ((int)threadIdx.x < w) { ss[threadIdx.x] += ss[threadIdx.x + w]; sq[threadIdx.x] += sq[threadIdx.x + w]; }
    __syncthreads();
  }
  if (threadIdx.x == 0) { sums[c] = ss[0]; sqs[c] = sq[0]; }
}

__global__ void k_bnapply(float* __restrict__ Y, int rows, int Opad, int O,
                          const float* __restrict__ sums, const float* __restrict__ sqs,
                          const float* __restrict__ g, const float* __restrict__ be,
                          int relu, int total)
{
  int i = blockIdx.x * blockDim.x + threadIdx.x;
  if (i >= total) return;
  int c = i % Opad; if (c >= O) return;
  float inv = 1.f / (float)rows;
  float m = sums[c] * inv;
  float v = sqs[c] * inv - m * m;
  float x = (Y[i] - m) * rsqrtf(v + 1e-5f) * g[c] + be[c];
  if (relu) x = fmaxf(x, 0.f);
  Y[i] = x;
}

__global__ void k_maxpool(const float* __restrict__ Y, float* __restrict__ out,
                          int BS, int O, int Opad, int total)
{
  int i = blockIdx.x * blockDim.x + threadIdx.x;
  if (i >= total) return;
  int p = i / O, c = i % O;
  const float* base = Y + (size_t)p * KK * Opad + c;
  float m = base[0];
  for (int k = 1; k < KK; ++k) m = fmaxf(m, base[(size_t)k * Opad]);
  out[(size_t)p * O + c] = m;
}

// [B,C,N] -> [B,N,C]
__global__ void k_t_cn_nc(const float* __restrict__ in, float* __restrict__ out,
                          int C, int Nn, int total)
{
  int i = blockIdx.x * blockDim.x + threadIdx.x;
  if (i >= total) return;
  int n = i % Nn; int t = i / Nn; int c = t % C; int b = t / C;
  out[((size_t)(b * Nn + n)) * C + c] = in[i];
}

// [B,Nn,C] (row stride) -> [B,C,Nn]
__global__ void k_out_nc_cn(const float* __restrict__ in, int stride,
                            float* __restrict__ out, int Nn, int C, int total)
{
  int i = blockIdx.x * blockDim.x + threadIdx.x;
  if (i >= total) return;
  int c = i % C; int bn = i / C; int n = bn % Nn; int b = bn / Nn;
  out[((size_t)(b * C + c)) * Nn + n] = in[(size_t)bn * stride + c];
}

__global__ void k_copycols(const float* __restrict__ Y, int stride,
                           float* __restrict__ out, int C, int total)
{
  int i = blockIdx.x * blockDim.x + threadIdx.x;
  if (i >= total) return;
  int r = i / C, c = i % C;
  out[i] = Y[(size_t)r * stride + c];
}

// farthest point sampling, one block (256 thr) per batch
__global__ void k_fps(const float* __restrict__ xyzT, int Nin, int S,
                      int* __restrict__ fidx, float* __restrict__ dist)
{
  int b = blockIdx.x; int t = threadIdx.x; const int T = 256;
  const float* P = xyzT + (size_t)b * Nin * 3;
  float* D = dist + (size_t)b * Nin;
  for (int i = t; i < Nin; i += T) D[i] = 1e10f;
  __shared__ float sv[256]; __shared__ int si[256];
  int far = 0;
  for (int it = 0; it < S; ++it) {
    if (t == 0) fidx[b * S + it] = far;
    float cx = P[(size_t)far*3], cy = P[(size_t)far*3+1], cz = P[(size_t)far*3+2];
    float best = -1.f; int bi = 0;
    for (int i = t; i < Nin; i += T) {
      float dx = P[(size_t)i*3] - cx, dy = P[(size_t)i*3+1] - cy, dz = P[(size_t)i*3+2] - cz;
      float d = dx*dx + dy*dy + dz*dz;
      float nd = fminf(D[i], d); D[i] = nd;
      if (nd > best) { best = nd; bi = i; }
    }
    sv[t] = best; si[t] = bi; __syncthreads();
    for (int w = 128; w > 0; w >>= 1) {
      if (t < w) {
        if (sv[t+w] > sv[t] || (sv[t+w] == sv[t] && si[t+w] < si[t])) { sv[t] = sv[t+w]; si[t] = si[t+w]; }
      }
      __syncthreads();
    }
    far = si[0];
    __syncthreads();
  }
}

__global__ void k_gather3(const float* __restrict__ xyzT, const int* __restrict__ fidx,
                          float* __restrict__ nx, int Nin, int S, int total)
{
  int i = blockIdx.x * blockDim.x + threadIdx.x;
  if (i >= total) return;
  int b = i / S; int f = fidx[i];
  nx[(size_t)i*3+0] = xyzT[((size_t)b*Nin + f)*3 + 0];
  nx[(size_t)i*3+1] = xyzT[((size_t)b*Nin + f)*3 + 1];
  nx[(size_t)i*3+2] = xyzT[((size_t)b*Nin + f)*3 + 2];
}

// ball query: first KK indices (ascending) with d^2 <= r^2, pad with first
__global__ void k_ball(const float* __restrict__ xyzT, const float* __restrict__ nx,
                       int* __restrict__ gidx, int Nin, int S, float r2, int total)
{
  int i = blockIdx.x * blockDim.x + threadIdx.x;
  if (i >= total) return;
  int b = i / S;
  float cx = nx[(size_t)i*3], cy = nx[(size_t)i*3+1], cz = nx[(size_t)i*3+2];
  const float* P = xyzT + (size_t)b * Nin * 3;
  int* out = gidx + (size_t)i * KK;
  int cnt = 0, first = Nin - 1; bool got = false;
  for (int j = 0; j < Nin; ++j) {
    float dx = P[(size_t)j*3] - cx, dy = P[(size_t)j*3+1] - cy, dz = P[(size_t)j*3+2] - cz;
    if (dx*dx + dy*dy + dz*dz <= r2) {
      if (!got) { got = true; first = j; }
      out[cnt++] = j;
      if (cnt == KK) break;
    }
  }
  for (; cnt < KK; ++cnt) out[cnt] = first;
}

// build grouped input rows: [B*S*K, Cpad] = [centered xyz(3) | gathered feat(Din) | 0 pad]
__global__ void k_group(const float* __restrict__ xyzT, const float* __restrict__ featT,
                        const float* __restrict__ nx, const int* __restrict__ gidx,
                        float* __restrict__ X, int Nin, int S, int Din, int Cpad, int total)
{
  int i = blockIdx.x * blockDim.x + threadIdx.x;
  if (i >= total) return;
  int c = i % Cpad; int t = i / Cpad; int k = t % KK; int bs = t / KK; int b = bs / S;
  int j = gidx[(size_t)bs * KK + k];
  float v = 0.f;
  if (c < 3) v = xyzT[((size_t)b*Nin + j)*3 + c] - nx[(size_t)bs*3 + c];
  else if (c - 3 < Din) v = featT[((size_t)b*Nin + j)*Din + (c - 3)];
  X[i] = v;
}

// 3 nearest neighbors of x1 points in x2, inverse-distance weights
__global__ void k_knn3(const float* __restrict__ x1, const float* __restrict__ x2,
                       int N1, int N2, int* __restrict__ idx3, float* __restrict__ w3, int total)
{
  int i = blockIdx.x * blockDim.x + threadIdx.x;
  if (i >= total) return;
  int b = i / N1;
  float px = x1[(size_t)i*3], py = x1[(size_t)i*3+1], pz = x1[(size_t)i*3+2];
  const float* Q = x2 + (size_t)b * N2 * 3;
  float d0 = 3e38f, d1 = 3e38f, d2 = 3e38f; int i0 = 0, i1 = 0, i2 = 0;
  for (int j = 0; j < N2; ++j) {
    float dx = Q[(size_t)j*3] - px, dy = Q[(size_t)j*3+1] - py, dz = Q[(size_t)j*3+2] - pz;
    float d = dx*dx + dy*dy + dz*dz;
    if (d < d0)      { d2=d1;i2=i1; d1=d0;i1=i0; d0=d;i0=j; }
    else if (d < d1) { d2=d1;i2=i1; d1=d;i1=j; }
    else if (d < d2) { d2=d;i2=j; }
  }
  float w0 = 1.f/(d0+1e-8f), w1 = 1.f/(d1+1e-8f), w2 = 1.f/(d2+1e-8f);
  float s = w0 + w1 + w2;
  w3[(size_t)i*3] = w0/s; w3[(size_t)i*3+1] = w1/s; w3[(size_t)i*3+2] = w2/s;
  idx3[(size_t)i*3] = i0; idx3[(size_t)i*3+1] = i1; idx3[(size_t)i*3+2] = i2;
}

// X[B*N1, Cpad] = [points1(C1) | interp(points2)(C2) | 0]
__global__ void k_interp(const float* __restrict__ p1, const float* __restrict__ p2,
                         const int* __restrict__ idx3, const float* __restrict__ w3,
                         float* __restrict__ X, int N1, int N2, int C1, int C2,
                         int Cpad, int total)
{
  int i = blockIdx.x * blockDim.x + threadIdx.x;
  if (i >= total) return;
  int c = i % Cpad; int bn = i / Cpad; int b = bn / N1;
  float v = 0.f;
  if (c < C1) v = p1[(size_t)bn * C1 + c];
  else if (c < C1 + C2) {
    int cc = c - C1;
    const float* P = p2 + (size_t)b * N2 * C2;
    const int* id = idx3 + (size_t)bn * 3;
    const float* w = w3 + (size_t)bn * 3;
    v = w[0]*P[(size_t)id[0]*C2+cc] + w[1]*P[(size_t)id[1]*C2+cc] + w[2]*P[(size_t)id[2]*C2+cc];
  }
  X[i] = v;
}

// fg/bg softmax over 2 channels -> mask
__global__ void k_mask(const float* __restrict__ Y2, int sY, float* __restrict__ mask, int total)
{
  int i = blockIdx.x * blockDim.x + threadIdx.x;
  if (i >= total) return;
  float l0 = Y2[(size_t)i*sY], l1 = Y2[(size_t)i*sY+1];
  float m = fmaxf(l0, l1);
  float e0 = expf(l0 - m), e1 = expf(l1 - m);
  float p1 = e1 / (e0 + e1);
  mask[i] = (p1 > 0.7f) ? 1.f : 0.f;
}

__global__ void k_vote(const float* __restrict__ xyzT, const float* __restrict__ feat,
                       const float* __restrict__ Yd, int sY, const float* __restrict__ mask,
                       float* __restrict__ vxyz, float* __restrict__ vfeat, int total)
{
  int i = blockIdx.x * blockDim.x + threadIdx.x;
  if (i >= total) return;
  int c = i % 131; int p = i / 131;
  float mk = mask[p];
  if (c < 3) vxyz[(size_t)p*3 + c] = (xyzT[(size_t)p*3 + c] + Yd[(size_t)p*sY + c]) * mk;
  else {
    int cc = c - 3;
    vfeat[(size_t)p*128 + cc] = (feat[(size_t)p*128 + cc] + Yd[(size_t)p*sY + 3 + cc]) * mk;
  }
}

__global__ void k_softmax11(const float* __restrict__ lg, float* __restrict__ pb, int rows)
{
  int i = blockIdx.x * blockDim.x + threadIdx.x;
  if (i >= rows) return;
  const float* x = lg + (size_t)i * 11; float* p = pb + (size_t)i * 11;
  float m = x[0];
  for (int c = 1; c < 11; ++c) m = fmaxf(m, x[c]);
  float e[11]; float s = 0.f;
  for (int c = 0; c < 11; ++c) { e[c] = expf(x[c] - m); s += e[c]; }
  for (int c = 0; c < 11; ++c) p[c] = e[c] / s;
}

__global__ void k_bsphere(const float* __restrict__ Y, int sY, const float* __restrict__ axyz,
                          float* __restrict__ out, int total)
{
  int i = blockIdx.x * blockDim.x + threadIdx.x;
  if (i >= total) return;
  int r = i / 4, c = i % 4;
  float v = Y[(size_t)r * sY + c];
  if (c < 3) v += axyz[(size_t)r*3 + c];
  out[i] = v;
}

__global__ void k_whl(const float* __restrict__ Y, int sY, const float* __restrict__ bsph,
                      float* __restrict__ out, int total)
{
  int i = blockIdx.x * blockDim.x + threadIdx.x;
  if (i >= total) return;
  int r = i / 3, c = i % 3;
  out[i] = Y[(size_t)r * sY + c] + 2.f * bsph[(size_t)r*4 + 3];
}

// ---------------------------------------------------------------------------
struct Layer {
  const float *W, *b, *g, *be;
  int Cin, Cout, Cpad, Opad;
  _Float16* Wh;
};

extern "C" void kernel_launch(void* const* d_in, const int* in_sizes, int n_in,
                              void* d_out, int out_size, void* d_ws, size_t ws_size,
                              hipStream_t stream)
{
  (void)in_sizes; (void)n_in; (void)out_size; (void)ws_size;
  auto in = [&](int i) -> const float* { return (const float*)d_in[i]; };

  char* ws = (char*)d_ws;
  size_t off = 0;
  auto alloc = [&](size_t bytes) -> void* {
    off = (off + 255) & ~(size_t)255;
    void* p = ws + off; off += bytes; return p;
  };

  // ---- persistent buffers ----
  float* xyzT = (float*)alloc((size_t)BB*NN*3*4);
  float* ptsT = (float*)alloc((size_t)BB*NN*2*4);
  float* nx1 = (float*)alloc((size_t)BB*4096*3*4); float* f1 = (float*)alloc((size_t)BB*4096*64*4);
  float* nx2 = (float*)alloc((size_t)BB*1024*3*4); float* f2 = (float*)alloc((size_t)BB*1024*128*4);
  float* nx3 = (float*)alloc((size_t)BB*256*3*4);  float* f3 = (float*)alloc((size_t)BB*256*256*4);
  float* nx4 = (float*)alloc((size_t)BB*64*3*4);   float* f4 = (float*)alloc((size_t)BB*64*512*4);
  float* fp4o = (float*)alloc((size_t)BB*256*256*4);
  float* fp3o = (float*)alloc((size_t)BB*1024*256*4);
  float* fp2o = (float*)alloc((size_t)BB*4096*128*4);
  float* feats = (float*)alloc((size_t)BB*NN*128*4);
  int*   fidx = (int*)alloc((size_t)BB*4096*4);
  int*   gidx = (int*)alloc((size_t)BB*4096*KK*4);
  float* dist = (float*)alloc((size_t)BB*NN*4);
  int*   idx3 = (int*)alloc((size_t)BB*NN*3*4);
  float* w3   = (float*)alloc((size_t)BB*NN*3*4);
  float* sums = (float*)alloc(512*4);
  float* sqs  = (float*)alloc(512*4);
  float* maskb = (float*)alloc((size_t)BB*NN*4);
  float* hx   = (float*)alloc((size_t)BB*NN*128*4);
  float* hlog = (float*)alloc((size_t)BB*NN*16*4);
  float* hdel = (float*)alloc((size_t)BB*NN*144*4);
  float* vx   = (float*)alloc((size_t)BB*NN*3*4);
  float* vf   = (float*)alloc((size_t)BB*NN*128*4);
  float* nx5  = (float*)alloc((size_t)BB*128*3*4);
  float* f5   = (float*)alloc((size_t)BB*128*128*4);
  float* h5   = (float*)alloc((size_t)256*128*4);
  float* hy[5];
  for (int i = 0; i < 5; ++i) hy[i] = (float*)alloc((size_t)256*16*4);
  _Float16* bufH = (_Float16*)alloc((size_t)17825792);  // f16 activations (max rows*Cpad)
  float* bufA = (float*)alloc((size_t)33554432);        // 32 MB
  float* bufB = (float*)alloc((size_t)67108864);        // 64 MB

  // ---- param layers (JAX pytree order: dict keys sorted alphabetically) ----
  // 0:xyz 1:points 2-3:bn1 4-5:bn3 6-7:bn5 8-9:conv1 10-11:conv2 12-13:conv3
  // 14-15:conv4 16-17:conv5 18-19:conv_bsphere 20-21:conv_class 22-23:conv_velocity
  // 24-25:conv_whl 26-27:conv_yaw 28..39:fp1 40..47:fp2 48..55:fp3 56..63:fp4
  // 64..75:sa1 76..87:sa2 88..99:sa3 100..111:sa4 112..123:vote
  auto prep = [&](Layer& L) {
    size_t n = (size_t)L.Opad * L.Cpad;
    L.Wh = (_Float16*)alloc(n * 2);
    k_wcvt<<<nblk(n, 256), 256, 0, stream>>>(L.W, L.Wh, L.Cout, L.Cin, L.Opad, L.Cpad);
  };
  auto fill = [&](Layer* Ls, int nl, int base, const int* ch) {
    for (int i = 0; i < nl; ++i) {
      Ls[i].W = in(base + 4*i);     Ls[i].b  = in(base + 4*i + 1);
      Ls[i].g = in(base + 4*i + 2); Ls[i].be = in(base + 4*i + 3);
      Ls[i].Cin = ch[i]; Ls[i].Cout = ch[i+1];
      Ls[i].Cpad = cp32(ch[i]); Ls[i].Opad = cp32(ch[i+1]);
      prep(Ls[i]);
    }
  };
  auto fillconv = [&](Layer& L, int baseW, int Cin, int Cout) {
    L.W = in(baseW); L.b = in(baseW + 1); L.g = nullptr; L.be = nullptr;
    L.Cin = Cin; L.Cout = Cout; L.Cpad = cp32(Cin); L.Opad = cp16(Cout);
    prep(L);
  };

  Layer Lsa1[3], Lsa2[3], Lsa3[3], Lsa4[3], Lvot[3], Lfp1[3], Lfp2[2], Lfp3[2], Lfp4[2];
  const int c_sa1[4] = {5,32,32,64};
  const int c_sa2[4] = {67,64,64,128};
  const int c_sa3[4] = {131,128,128,256};
  const int c_sa4[4] = {259,256,256,512};
  const int c_fp4[3] = {768,256,256};
  const int c_fp3[3] = {384,256,256};
  const int c_fp2[3] = {320,256,128};
  const int c_fp1[4] = {128,128,128,128};
  const int c_vot[4] = {131,128,128,128};
  fill(Lfp1, 3, 28, c_fp1);  fill(Lfp2, 2, 40, c_fp2);
  fill(Lfp3, 2, 48, c_fp3);  fill(Lfp4, 2, 56, c_fp4);
  fill(Lsa1, 3, 64, c_sa1);  fill(Lsa2, 3, 76, c_sa2);
  fill(Lsa3, 3, 88, c_sa3);  fill(Lsa4, 3, 100, c_sa4);
  fill(Lvot, 3, 112, c_vot);
  Layer Lc1, Lc2, Lc3, Lc4, Lc5, Lcls, Lbs, Lwhl, Lyaw, Lvel;
  fillconv(Lc1, 8, 128, 128);  fillconv(Lc2, 10, 128, 2);
  fillconv(Lc3, 12, 128, 128); fillconv(Lc4, 14, 128, 131);
  fillconv(Lc5, 16, 128, 128); fillconv(Lbs, 18, 128, 4);
  fillconv(Lcls, 20, 128, 11); fillconv(Lvel, 22, 128, 2);
  fillconv(Lwhl, 24, 128, 3);  fillconv(Lyaw, 26, 128, 1);

  // convert activations to f16 then run WMMA GEMM
  auto gemm = [&](const float* X, Layer& L, float* Y, int rows) {
    int tot = rows * L.Cpad;
    k_cvt_h<<<nblk((size_t)tot, 256), 256, 0, stream>>>(X, bufH, tot);
    dim3 g(nblk((size_t)(rows / 16), 4), (unsigned)(L.Opad / 16));
    k_gemm<<<g, 128, 0, stream>>>(bufH, L.Wh, L.b, Y, rows, L.Cpad, L.Cout, L.Opad);
  };

  // ---- generic layer runner (GEMM -> BN stats -> BN+ReLU) with A/B ping-pong ----
  auto run_layers = [&](Layer* L, int nl, int rows, float* X, float* finalOut) -> float* {
    float* bufs[2] = { bufA, bufB };
    float* cur = X;
    int w = (X == bufA) ? 1 : 0;
    for (int l = 0; l < nl; ++l) {
      int O = L[l].Cout, Op = L[l].Opad;
      float* Y = (l == nl - 1 && finalOut) ? finalOut : bufs[w];
      w ^= 1;
      gemm(cur, L[l], Y, rows);
      k_bnstat<<<O, 256, 0, stream>>>(Y, rows, Op, sums, sqs, O);
      int tot = rows * Op;
      k_bnapply<<<nblk((size_t)tot, 256), 256, 0, stream>>>(Y, rows, Op, O, sums, sqs,
                                                            L[l].g, L[l].be, 1, tot);
      cur = Y;
    }
    return cur;
  };

  auto sa_stage = [&](const float* pxyz, int Nin, const float* pfeat, int Din,
                      int S, float radius, Layer* L, int nl, float* nx, float* pf) {
    k_fps<<<BB, 256, 0, stream>>>(pxyz, Nin, S, fidx, dist);
    k_gather3<<<nblk((size_t)BB*S, 256), 256, 0, stream>>>(pxyz, fidx, nx, Nin, S, BB*S);
    k_ball<<<nblk((size_t)BB*S, 256), 256, 0, stream>>>(pxyz, nx, gidx, Nin, S,
                                                        radius*radius, BB*S);
    int Cp = cp32(3 + Din);
    int rows = BB * S * KK;
    int tot = rows * Cp;
    k_group<<<nblk((size_t)tot, 256), 256, 0, stream>>>(pxyz, pfeat, nx, gidx, bufA,
                                                        Nin, S, Din, Cp, tot);
    float* Y = run_layers(L, nl, rows, bufA, nullptr);
    int Cout = L[nl-1].Cout;
    int pt = BB * S * Cout;
    k_maxpool<<<nblk((size_t)pt, 256), 256, 0, stream>>>(Y, pf, BB*S, Cout, L[nl-1].Opad, pt);
  };

  auto fp_stage = [&](const float* x1, int N1, const float* x2, int N2,
                      const float* p1, int C1, const float* p2, int C2,
                      Layer* L, int nl, float* outb) {
    k_knn3<<<nblk((size_t)BB*N1, 256), 256, 0, stream>>>(x1, x2, N1, N2, idx3, w3, BB*N1);
    int Cp = cp32(C1 + C2);
    int rows = BB * N1;
    int tot = rows * Cp;
    k_interp<<<nblk((size_t)tot, 256), 256, 0, stream>>>(p1, p2, idx3, w3, bufA,
                                                         N1, N2, C1, C2, Cp, tot);
    run_layers(L, nl, rows, bufA, outb);
  };

  // ---- backbone ----
  k_t_cn_nc<<<nblk((size_t)BB*3*NN, 256), 256, 0, stream>>>(in(0), xyzT, 3, NN, BB*3*NN);
  k_t_cn_nc<<<nblk((size_t)BB*2*NN, 256), 256, 0, stream>>>(in(1), ptsT, 2, NN, BB*2*NN);

  sa_stage(xyzT, NN,   ptsT, 2,   4096, 1.0f, Lsa1, 3, nx1, f1);
  sa_stage(nx1,  4096, f1,   64,  1024, 2.0f, Lsa2, 3, nx2, f2);
  sa_stage(nx2,  1024, f2,   128, 256,  4.0f, Lsa3, 3, nx3, f3);
  sa_stage(nx3,  256,  f3,   256, 64,   8.0f, Lsa4, 3, nx4, f4);

  fp_stage(nx3, 256,  nx4, 64,   f3, 256, f4,   512, Lfp4, 2, fp4o);
  fp_stage(nx2, 1024, nx3, 256,  f2, 128, fp4o, 256, Lfp3, 2, fp3o);
  fp_stage(nx1, 4096, nx2, 1024, f1, 64,  fp3o, 256, Lfp2, 2, fp2o);
  fp_stage(xyzT, NN,  nx1, 4096, nullptr, 0, fp2o, 128, Lfp1, 3, feats);

  // ---- output layout (flat f32, reference return order) ----
  float* dout = (float*)d_out;
  float* o_feat = dout;                    // [B,128,N]  2097152
  float* o_log  = o_feat + 2097152;        // [B,2,N]    32768
  float* o_del  = o_log + 32768;           // [B,N,3]    49152
  float* o_agg  = o_del + 49152;           // [B,128,3]  768
  float* o_cls  = o_agg + 768;             // [B,128,11] 2816
  float* o_clsp = o_cls + 2816;            // [B,128,11] 2816
  float* o_bs   = o_clsp + 2816;           // [B,128,4]  1024
  float* o_whl  = o_bs + 1024;             // [B,128,3]  768
  float* o_yaw  = o_whl + 768;             // [B,128,1]  256
  float* o_vel  = o_yaw + 256;             // [B,128,2]  512

  k_out_nc_cn<<<nblk((size_t)BB*NN*128, 256), 256, 0, stream>>>(feats, 128, o_feat,
                                                                NN, 128, BB*NN*128);

  int rowsN = BB * NN;
  // conv1 + bn1 + relu -> hx
  gemm(feats, Lc1, hx, rowsN);
  k_bnstat<<<128, 256, 0, stream>>>(hx, rowsN, 128, sums, sqs, 128);
  k_bnapply<<<nblk((size_t)rowsN*128, 256), 256, 0, stream>>>(hx, rowsN, 128, 128, sums, sqs,
                                                              in(2), in(3), 1, rowsN*128);
  // conv2 -> hlog ; logits out ; mask
  gemm(hx, Lc2, hlog, rowsN);
  k_out_nc_cn<<<nblk((size_t)BB*NN*2, 256), 256, 0, stream>>>(hlog, 16, o_log, NN, 2, BB*NN*2);
  k_mask<<<nblk((size_t)rowsN, 256), 256, 0, stream>>>(hlog, 16, maskb, rowsN);
  // conv3 + bn3 + relu -> hx
  gemm(feats, Lc3, hx, rowsN);
  k_bnstat<<<128, 256, 0, stream>>>(hx, rowsN, 128, sums, sqs, 128);
  k_bnapply<<<nblk((size_t)rowsN*128, 256), 256, 0, stream>>>(hx, rowsN, 128, 128, sums, sqs,
                                                              in(4), in(5), 1, rowsN*128);
  // conv4 -> hdel ; delta out ; vote inputs
  gemm(hx, Lc4, hdel, rowsN);
  k_copycols<<<nblk((size_t)rowsN*3, 256), 256, 0, stream>>>(hdel, 144, o_del, 3, rowsN*3);
  k_vote<<<nblk((size_t)rowsN*131, 256), 256, 0, stream>>>(xyzT, feats, hdel, 144, maskb,
                                                           vx, vf, rowsN*131);
  // vote set-abstraction
  sa_stage(vx, NN, vf, 128, 128, 4.0f, Lvot, 3, nx5, f5);
  k_copycols<<<nblk((size_t)BB*128*3, 256), 256, 0, stream>>>(nx5, 3, o_agg, 3, BB*128*3);

  // conv5 + bn5 + relu -> h5
  int rows5 = BB * 128;
  gemm(f5, Lc5, h5, rows5);
  k_bnstat<<<128, 256, 0, stream>>>(h5, rows5, 128, sums, sqs, 128);
  k_bnapply<<<nblk((size_t)rows5*128, 256), 256, 0, stream>>>(h5, rows5, 128, 128, sums, sqs,
                                                              in(6), in(7), 1, rows5*128);
  // heads
  gemm(h5, Lcls, hy[0], rows5);
  gemm(h5, Lbs,  hy[1], rows5);
  gemm(h5, Lwhl, hy[2], rows5);
  gemm(h5, Lyaw, hy[3], rows5);
  gemm(h5, Lvel, hy[4], rows5);
  k_copycols<<<nblk((size_t)rows5*11, 256), 256, 0, stream>>>(hy[0], 16, o_cls, 11, rows5*11);
  k_softmax11<<<nblk((size_t)rows5, 256), 256, 0, stream>>>(o_cls, o_clsp, rows5);
  k_bsphere<<<nblk((size_t)rows5*4, 256), 256, 0, stream>>>(hy[1], 16, nx5, o_bs, rows5*4);
  k_whl<<<nblk((size_t)rows5*3, 256), 256, 0, stream>>>(hy[2], 16, o_bs, o_whl, rows5*3);
  k_copycols<<<nblk((size_t)rows5*1, 256), 256, 0, stream>>>(hy[3], 16, o_yaw, 1, rows5*1);
  k_copycols<<<nblk((size_t)rows5*2, 256), 256, 0, stream>>>(hy[4], 16, o_vel, 2, rows5*2);
}
